// BaselineMLP_38680475468244
// MI455X (gfx1250) — matrix operand order repaired and checked by reading, not compiled, not run on hardware
//
#include <hip/hip_runtime.h>

typedef _Float16 v16h __attribute__((ext_vector_type(16)));
typedef _Float16 v8h  __attribute__((ext_vector_type(8)));
typedef float    v8f  __attribute__((ext_vector_type(8)));

// Problem constants (match reference)
#define IN_DIM   256   // 2*N_FEAT
#define HID      256
#define HID2     128
#define NFEAT    128
#define EPT      64    // edges per workgroup tile (4 M-tiles of 16)

// LDS carve (bytes)
#define OFF_W1   0               // 256*256 f16 = 131072
#define OFF_W2   131072          // 128*256 f16 = 65536
#define OFF_A    196608          // 64*256 f16  = 32768
#define OFF_H1   229376          // 64*256 f16  = 32768
#define OFF_H2   262144          // 64*128 f16  = 16384
#define OFF_B1   278528          // 256 f32     = 1024
#define OFF_B2   279552          // 128 f32     = 512
#define OFF_W3   280064          // 128 f32     = 512
#define OFF_RED  280576          // 64*4 f32    = 1024
#define SMEM_BYTES 281600        // 275 KB < 320 KB WGP LDS

union V16U { v16h v; struct { v8h lo, hi; } s; };

// A fragment (16x32 f16): lane (l&15)=row; k chunks (l>>4)*8 and +16, 8 halfs each.
__device__ __forceinline__ v16h load_a_frag(const _Float16* base, int stride,
                                            int mtile, int kstep, int lane) {
    const int row = mtile * 16 + (lane & 15);
    const int k0  = kstep * 32 + ((lane >> 4) << 3);
    const _Float16* p = base + row * stride + k0;
    V16U u;
    u.s.lo = *(const v8h*)(p);
    u.s.hi = *(const v8h*)(p + 16);
    return u.v;
}

// B fragment (32x16 f16) from pre-transposed [N][K=256] weights:
// lane (l&15)=col; k chunks (l>>4)*8 and +16 (+ kstep*32), 8 halfs each.
__device__ __forceinline__ v16h load_b_frag(const _Float16* base,
                                            int ntile, int kstep, int lane) {
    const int col = ntile * 16 + (lane & 15);
    const int k0  = kstep * 32 + ((lane >> 4) << 3);
    const _Float16* p = base + col * 256 + k0;
    V16U u;
    u.s.lo = *(const v8h*)(p);
    u.s.hi = *(const v8h*)(p + 16);
    return u.v;
}

// One-time: convert + transpose weights to f16 [N][K] in workspace.
__global__ void prep_weights(const float* __restrict__ W1, const float* __restrict__ W2,
                             _Float16* __restrict__ W1t, _Float16* __restrict__ W2t) {
    const int b = blockIdx.x;
    const int t = threadIdx.x;            // 0..255
    if (b < HID) {                        // W1: [256 in][256 out] -> W1t[n][k]
        W1t[b * 256 + t] = (_Float16)W1[t * HID + b];
    } else {                              // W2: [256 in][128 out] -> W2t[n][k]
        const int n = b - HID;
        W2t[n * 256 + t] = (_Float16)W2[t * HID2 + n];
    }
}

__launch_bounds__(256)
__global__ void edge_mlp_kernel(const float* __restrict__ x,
                                const int* __restrict__ ei,
                                const _Float16* __restrict__ W1t,
                                const _Float16* __restrict__ W2t,
                                const float* __restrict__ b1,
                                const float* __restrict__ b2,
                                const float* __restrict__ W3,
                                const float* __restrict__ b3,
                                float* __restrict__ out,
                                int nEdges, int numTiles) {
    extern __shared__ char smem[];
    _Float16* W1s = (_Float16*)(smem + OFF_W1);
    _Float16* W2s = (_Float16*)(smem + OFF_W2);
    _Float16* Als = (_Float16*)(smem + OFF_A);
    _Float16* H1s = (_Float16*)(smem + OFF_H1);
    _Float16* H2s = (_Float16*)(smem + OFF_H2);
    float*    b1s = (float*)(smem + OFF_B1);
    float*    b2s = (float*)(smem + OFF_B2);
    float*    w3s = (float*)(smem + OFF_W3);
    float*    red = (float*)(smem + OFF_RED);

    const int tid  = threadIdx.x;
    const int lane = tid & 31;
    const int wave = tid >> 5;
    const float b3v = b3[0];

    // ---- stage all f16 weights into LDS (192 KB, L2-resident source) ----
    {
        const uint4* s1 = (const uint4*)W1t; uint4* d1 = (uint4*)W1s;
        for (int i = tid; i < (HID * 256) / 8; i += 256) d1[i] = s1[i];
        const uint4* s2 = (const uint4*)W2t; uint4* d2 = (uint4*)W2s;
        for (int i = tid; i < (HID2 * 256) / 8; i += 256) d2[i] = s2[i];
        b1s[tid] = b1[tid];
        if (tid < HID2) { b2s[tid] = b2[tid]; w3s[tid] = W3[tid]; }
    }
    __syncthreads();

    for (int tile = blockIdx.x; tile < numTiles; tile += gridDim.x) {
        const int tileBase = tile * EPT;

        // ---- gather: A[e][0:128]=x[row], A[e][128:256]=x[col], f32->f16 ----
        {
            const int g      = tid >> 1;        // 0..127: (edge, side)
            const int hsel   = tid & 1;         // which 64-float half
            const int eLocal = g >> 1;
            const int side   = g & 1;
            const int e      = tileBase + eLocal;
            _Float16* dst = Als + eLocal * IN_DIM + side * NFEAT + hsel * 64;
            if (e < nEdges) {
                const int node = (side == 0) ? ei[e] : ei[nEdges + e];
                const float4* s4 = (const float4*)(x + (long)node * NFEAT + hsel * 64);
#pragma unroll
                for (int i = 0; i < 16; ++i) {
                    float4 f = s4[i];
                    dst[i * 4 + 0] = (_Float16)f.x;
                    dst[i * 4 + 1] = (_Float16)f.y;
                    dst[i * 4 + 2] = (_Float16)f.z;
                    dst[i * 4 + 3] = (_Float16)f.w;
                }
            } else {
#pragma unroll
                for (int i = 0; i < 64; ++i) dst[i] = (_Float16)0.f;
            }
        }
        __syncthreads();

        // ---- layer 1: [64x256] = relu(A[64x256] @ W1 + b1) ----
        // wave -> M-tile (wave&3), 8 N-tiles starting at (wave>>2)*8
        {
            const int m  = wave & 3;
            const int nb = (wave >> 2) * 8;
            v8f acc[8] = {};
#pragma unroll
            for (int k = 0; k < 8; ++k) {
                v16h a = load_a_frag(Als, IN_DIM, m, k, lane);
#pragma unroll
                for (int j = 0; j < 8; ++j) {
                    v16h b = load_b_frag(W1s, nb + j, k, lane);
                    acc[j] = __builtin_amdgcn_wmma_f32_16x16x32_f16(
                        false, a, false, b, (short)0, acc[j], false, false);
                }
            }
            const int rowb = m * 16 + ((lane >> 4) << 3);
            const int cl   = lane & 15;
#pragma unroll
            for (int j = 0; j < 8; ++j) {
                const int n = (nb + j) * 16 + cl;
                const float bias = b1s[n];
#pragma unroll
                for (int r = 0; r < 8; ++r) {
                    float v = acc[j][r] + bias;
                    v = v > 0.f ? v : 0.f;                       // ReLU
                    H1s[(rowb + r) * HID + n] = (_Float16)v;
                }
            }
        }
        __syncthreads();

        // ---- layer 2: [64x128] = leaky_relu(H1[64x256] @ W2 + b2) ----
        // wave -> M-tile (wave&3), 4 N-tiles starting at (wave>>2)*4
        {
            const int m  = wave & 3;
            const int nb = (wave >> 2) * 4;
            v8f acc[4] = {};
#pragma unroll
            for (int k = 0; k < 8; ++k) {
                v16h a = load_a_frag(H1s, HID, m, k, lane);
#pragma unroll
                for (int j = 0; j < 4; ++j) {
                    v16h b = load_b_frag(W2s, nb + j, k, lane);
                    acc[j] = __builtin_amdgcn_wmma_f32_16x16x32_f16(
                        false, a, false, b, (short)0, acc[j], false, false);
                }
            }
            const int rowb = m * 16 + ((lane >> 4) << 3);
            const int cl   = lane & 15;
#pragma unroll
            for (int j = 0; j < 4; ++j) {
                const int n = (nb + j) * 16 + cl;
                const float bias = b2s[n];
#pragma unroll
                for (int r = 0; r < 8; ++r) {
                    float v = acc[j][r] + bias;
                    v = v > 0.f ? v : 0.01f * v;                 // LeakyReLU
                    H2s[(rowb + r) * HID2 + n] = (_Float16)v;
                }
            }
        }
        __syncthreads();

        // ---- layer 3: out[e] = H2[e] . W3 + b3  (4 lanes per edge) ----
        {
            const int e = tid >> 2;     // 0..63
            const int q = tid & 3;      // quarter of the 128 features
            const _Float16* hp = H2s + e * HID2 + q * 32;
            const float*    wp = w3s + q * 32;
            float sum = 0.f;
#pragma unroll
            for (int i = 0; i < 32; ++i) sum += (float)hp[i] * wp[i];
            red[e * 4 + q] = sum;
        }
        __syncthreads();
        if (tid < EPT) {
            const int eg = tileBase + tid;
            if (eg < nEdges) {
                out[eg] = red[tid * 4 + 0] + red[tid * 4 + 1] +
                          red[tid * 4 + 2] + red[tid * 4 + 3] + b3v;
            }
        }
        __syncthreads();   // protect LDS reuse next iteration
    }
}

extern "C" void kernel_launch(void* const* d_in, const int* in_sizes, int n_in,
                              void* d_out, int out_size, void* d_ws, size_t ws_size,
                              hipStream_t stream) {
    const float* x  = (const float*)d_in[0];
    const int*   ei = (const int*)d_in[1];
    const float* W1 = (const float*)d_in[2];
    const float* b1 = (const float*)d_in[3];
    const float* W2 = (const float*)d_in[4];
    const float* b2 = (const float*)d_in[5];
    const float* W3 = (const float*)d_in[6];
    const float* b3 = (const float*)d_in[7];
    float* out = (float*)d_out;

    const int nEdges = in_sizes[1] / 2;

    _Float16* W1t = (_Float16*)d_ws;                        // 128 KB
    _Float16* W2t = (_Float16*)((char*)d_ws + 131072);      // 64 KB

    prep_weights<<<HID + HID2, 256, 0, stream>>>(W1, W2, W1t, W2t);

    const int numTiles = (nEdges + EPT - 1) / EPT;          // 9375 for E=600k
    int grid = numTiles < 512 ? numTiles : 512;             // persistent-ish blocks
    edge_mlp_kernel<<<grid, 256, SMEM_BYTES, stream>>>(
        x, ei, W1t, W2t, b1, b2, W3, b3, out, nEdges, numTiles);
}
